// SuperpixelAttention_87110526697933
// MI455X (gfx1250) — matrix operand order repaired and checked by reading, not compiled, not run on hardware
//
#include <hip/hip_runtime.h>
#include <hip/hip_bf16.h>

typedef _Float16 v16h __attribute__((ext_vector_type(16)));
typedef _Float16 v8h  __attribute__((ext_vector_type(8)));
typedef float    v8f  __attribute__((ext_vector_type(8)));
typedef unsigned int v4u __attribute__((ext_vector_type(4)));
typedef int          v8i __attribute__((ext_vector_type(8)));
typedef int          v4i __attribute__((ext_vector_type(4)));

#define TT   4
#define CC   256
#define HH   256
#define WWD  256
#define SS   9
#define NHD  4
#define HDIM 64
#define HW   (HH*WWD)          // 65536
#define NPIX (TT*HH*WWD)       // 262144
#define KSTR 72                // padded per-pixel f16 stride in attention LDS
#define WPITCH 264             // padded f16 row pitch for LDS-staged weights (256+8)

#if defined(__has_builtin)
#if __has_builtin(__builtin_amdgcn_tensor_load_to_lds) && __has_builtin(__builtin_amdgcn_s_wait_tensorcnt)
#define HAVE_TDM 1
#endif
#endif

static __device__ __forceinline__ v16h cat16(v8h lo, v8h hi) {
    v16h r;
#pragma unroll
    for (int i = 0; i < 8; ++i) { r[i] = lo[i]; r[8 + i] = hi[i]; }
    return r;
}
static __device__ __forceinline__ int iclamp(int v, int hi) {
    return v < 0 ? 0 : (v > hi ? hi : v);
}

// Stage one 256x256 f16 weight (row-major, row stride 256) into LDS with a
// 264-half padded pitch.  TDM path: one wave issues a 2D tensor DMA with LDS
// padding (pad every 512B by 16B); fallback: cooperative b128 copies.
static __device__ __forceinline__ void stage_weight_lds(
        _Float16* wlds, const _Float16* src, int wave, int tid) {
#ifdef HAVE_TDM
    if (wave == 0) {
        const unsigned long long ga = (unsigned long long)(uintptr_t)src;
        const unsigned loff = (unsigned)(uintptr_t)wlds;  // LDS byte offset
        v4u g0;
        g0[0] = 1u;                                        // count=1, user mode
        g0[1] = loff;                                      // lds_addr
        g0[2] = (unsigned)(ga & 0xffffffffu);              // global_addr[31:0]
        g0[3] = (unsigned)((ga >> 32) & 0x01ffffffu)       // global_addr[56:32]
              | (2u << 30);                                // type=2 ("image")
        v8i g1;
        g1[0] = (int)((1u << 16)      // data_size = 1 -> 2 bytes
                    | (1u << 20)      // pad_enable
                    | (6u << 22)      // pad_interval: 2^(6+1)=128 DW = 512B
                    | (3u << 25));    // pad_amount: 4 DW = 16B  (-> 264h pitch)
        g1[1] = (int)(256u << 16);    // tensor_dim0[15:0] @ bits 63:48
        g1[2] = (int)(256u << 16);    // tensor_dim1[15:0] @ bits 95:80
        g1[3] = (int)(256u << 16);    // tile_dim0 @ bits 127:112
        g1[4] = 256;                  // tile_dim1 @ bits 143:128 (tile_dim2=0)
        g1[5] = 256;                  // tensor_dim0_stride[31:0]
        g1[6] = 0;                    // tds0[47:32]=0, tds1[15:0]=0
        g1[7] = 1;                    // tds1[47:16] (=65536; unused for 2D)
        v4i g2 = {0, 0, 0, 0};        // D# group 2 (unused for 2D tile)
        v4i g3 = {0, 0, 0, 0};        // D# group 3 (unused for 2D tile)
        v8i g4 = {0, 0, 0, 0, 0, 0, 0, 0};   // extra words (clang-23 6-arg form)
        __builtin_amdgcn_tensor_load_to_lds(g0, g1, g2, g3, g4, 0);
        __builtin_amdgcn_s_wait_tensorcnt(0);
    }
#else
    for (int u = tid; u < 256 * 32; u += 256) {
        const int row = u >> 5, ch = u & 31;
        *(v8h*)(wlds + row * WPITCH + ch * 8) =
            *(const v8h*)(src + row * 256 + ch * 8);
    }
#endif
    __syncthreads();
}

// ---------------------------------------------------------------------------
// K0: convert the four 256x256 f32 weights into f16, transposed: Wt[n][k]=W[k][n]
// ---------------------------------------------------------------------------
__global__ void __launch_bounds__(256) k_wt(const float* __restrict__ Wq,
                                            const float* __restrict__ Wk,
                                            const float* __restrict__ Wv,
                                            const float* __restrict__ Wp,
                                            _Float16* __restrict__ wt) {
    const int k = blockIdx.x;        // 0..255 (input row -> coalesced read)
    const int mat = blockIdx.y;      // 0..3
    const int n = threadIdx.x;       // 0..255
    const float* src = (mat == 0) ? Wq : (mat == 1) ? Wk : (mat == 2) ? Wv : Wp;
    wt[mat * 65536 + n * 256 + k] = (_Float16)src[k * 256 + n];
}

// ---------------------------------------------------------------------------
// K1: fused QKV projection.  Weights staged per-workgroup into LDS via the
// Tensor Data Mover; each wave owns a 16-pixel row tile with its A fragments
// resident in VGPRs, and sweeps 3 weights x 16 col-tiles x 8 k-steps of
// v_wmma_f32_16x16x32_f16 fed from conflict-free ds_load_b128.
// ---------------------------------------------------------------------------
__global__ void __launch_bounds__(256) k_qkv(const float* __restrict__ x,
                                             const _Float16* __restrict__ wt,
                                             _Float16* __restrict__ qh,
                                             _Float16* __restrict__ kh,
                                             _Float16* __restrict__ vh) {
    __shared__ _Float16 wlds[256 * WPITCH];   // 135,168 B
    const int tid  = threadIdx.x;
    const int lane = tid & 31;
    const int wave = tid >> 5;
    const int rt = blockIdx.x * 8 + wave;      // 16-row tile index
    const int pixbase = rt * 16;
    const int t  = pixbase >> 16;
    const int h  = (pixbase >> 8) & 255;
    const int wb = pixbase & 255;
    const int ln  = lane & 15;
    const int ak8 = (lane >> 4) << 3;          // A-frag K sub-block (0 or 8)
    const int bk16 = (lane >> 4) << 3;         // B-frag K sub-block (0 or 8)

    // Preload all 8 A fragments (K = 256 in steps of 32).
    // ISA 16-bit A layout: lane<16 holds K = k0..k0+7 (halves 0..7) and
    // k0+16..k0+23 (halves 8..15) with k0 = ks*32 (+8 for lanes 16..31).
    v16h afr[8];
    {
        const float* xb = x + (size_t)t * CC * HW + h * WWD + wb + ln;
#pragma unroll
        for (int ks = 0; ks < 8; ++ks) {
            const float* px = xb + (size_t)(ks * 32 + ak8) * HW;
            v16h a;
#pragma unroll
            for (int j = 0; j < 8; ++j) {
                a[j]     = (_Float16)px[(size_t)j * HW];
                a[8 + j] = (_Float16)px[(size_t)(j + 16) * HW];
            }
            afr[ks] = a;
        }
    }

    for (int m = 0; m < 3; ++m) {
        __syncthreads();   // protect wlds reuse across weight iterations
        stage_weight_lds(wlds, wt + m * 65536, wave, tid);

        _Float16* dst = (m == 0) ? qh : (m == 1) ? kh : vh;
        const float scl = (m == 0) ? 0.125f : 1.0f;   // q * hd^-0.5
        for (int ct = 0; ct < 16; ++ct) {
            v8f acc = {};
            // B layout: lane = column N, K = k0..k0+7 and k0+16..k0+23 (k0 += 8 for lanes 16..31).
            const _Float16* wrow = wlds + (ct * 16 + ln) * WPITCH + bk16;
#pragma unroll
            for (int ks = 0; ks < 8; ++ks) {
                v8h lo = *(const v8h*)(wrow + ks * 32);
                v8h hi = *(const v8h*)(wrow + ks * 32 + 16);
                v16h b = cat16(lo, hi);
                acc = __builtin_amdgcn_wmma_f32_16x16x32_f16(
                        false, afr[ks], false, b, (short)0, acc, false, false);
            }
            // C layout: VGPR r -> row r (lanes 0-15) / r+8 (lanes 16-31)
#pragma unroll
            for (int r = 0; r < 8; ++r) {
                const int mrow = r + ak8;
                dst[(size_t)(pixbase + mrow) * 256 + ct * 16 + ln] =
                    (_Float16)(acc[r] * scl);
            }
        }
    }
}

// ---------------------------------------------------------------------------
// K2: windowed superpixel attention.  One block = 16x16 pixel tile of one t.
// LDS stages the 20x20 neighborhood of k/v (per head) and sims; border
// clamping (jnp.clip semantics) is done at load, which precludes a plain
// rectangular TDM tile here.
// ---------------------------------------------------------------------------
__global__ void __launch_bounds__(256) k_attn(const _Float16* __restrict__ qh,
                                              const _Float16* __restrict__ kh,
                                              const _Float16* __restrict__ vh,
                                              const float* __restrict__ sims,
                                              _Float16* __restrict__ ah) {
    __shared__ _Float16 kt[400 * KSTR];   // 57,600 B
    __shared__ _Float16 vt[400 * KSTR];   // 57,600 B
    __shared__ float    st[400 * SS];     // 14,400 B  (total 129,600 B < 320 KB)

    const int t  = blockIdx.z;
    const int h0 = blockIdx.y * 16;
    const int w0 = blockIdx.x * 16;
    const int tid = threadIdx.x;

    // Stage sims neighborhood (clipped at borders).
    for (int i = tid; i < 400; i += 256) {
        const int ly = i / 20, lx = i % 20;
        const int gh = iclamp(h0 + ly - 2, HH - 1);
        const int gw = iclamp(w0 + lx - 2, WWD - 1);
        const float* sp = sims + (size_t)t * SS * HW + gh * WWD + gw;
#pragma unroll
        for (int s = 0; s < SS; ++s) st[i * SS + s] = sp[(size_t)s * HW];
    }
    __syncthreads();

    const int py = tid >> 4, px = tid & 15;
    const int ly = py + 2,  lx = px + 2;
    const int mypix = t * HW + (h0 + py) * WWD + (w0 + px);

    // Superpixel similarity weights (shared across heads).
    float msim[SS];
#pragma unroll
    for (int s = 0; s < SS; ++s) msim[s] = st[(ly * 20 + lx) * SS + s];
    float simw[25];
#pragma unroll
    for (int kk = 0; kk < 25; ++kk) {
        const int di = kk / 5 - 2, dj = kk % 5 - 2;
        const float* sp = st + ((ly + di) * 20 + (lx + dj)) * SS;
        float s = 0.f;
#pragma unroll
        for (int j = 0; j < SS; ++j) s += msim[j] * sp[j];
        simw[kk] = s;
    }

    for (int hd = 0; hd < NHD; ++hd) {
        __syncthreads();   // protect kt/vt reuse across head iterations
        // Stage k/v 20x20 neighborhood for this head (64 halves per pixel).
        for (int u = tid; u < 400 * 8; u += 256) {
            const int pix = u >> 3, ch = u & 7;
            const int ty = pix / 20, tx = pix % 20;
            const int gh = iclamp(h0 + ty - 2, HH - 1);
            const int gw = iclamp(w0 + tx - 2, WWD - 1);
            const size_t goff =
                (size_t)(t * HW + gh * WWD + gw) * 256 + hd * HDIM + ch * 8;
            *(v8h*)(kt + pix * KSTR + ch * 8) = *(const v8h*)(kh + goff);
            *(v8h*)(vt + pix * KSTR + ch * 8) = *(const v8h*)(vh + goff);
        }
        __syncthreads();

        // Own q vector (already scaled).
        float qf[HDIM];
        {
            const _Float16* qp = qh + (size_t)mypix * 256 + hd * HDIM;
#pragma unroll
            for (int c8 = 0; c8 < 8; ++c8) {
                v8h q8 = *(const v8h*)(qp + c8 * 8);
#pragma unroll
                for (int j = 0; j < 8; ++j) qf[c8 * 8 + j] = (float)q8[j];
            }
        }

        // Logits over the 25 offsets.
        float lg[25];
#pragma unroll
        for (int kk = 0; kk < 25; ++kk) {
            const int di = kk / 5 - 2, dj = kk % 5 - 2;
            const _Float16* kp = kt + ((ly + di) * 20 + (lx + dj)) * KSTR;
            float s = 0.f;
#pragma unroll
            for (int c8 = 0; c8 < 8; ++c8) {
                v8h k8 = *(const v8h*)(kp + c8 * 8);
#pragma unroll
                for (int j = 0; j < 8; ++j) s += qf[c8 * 8 + j] * (float)k8[j];
            }
            lg[kk] = s;
        }

        // exp-shift, reweight by sims, renormalize.
        float mx = lg[0];
#pragma unroll
        for (int kk = 1; kk < 25; ++kk) mx = fmaxf(mx, lg[kk]);
        float ssum = 0.f;
#pragma unroll
        for (int kk = 0; kk < 25; ++kk) {
            lg[kk] = __expf(lg[kk] - mx) * simw[kk];
            ssum += lg[kk];
        }
        const float inv = 1.0f / (1e-10f + ssum);

        // Weighted V accumulation.
        float o[HDIM];
#pragma unroll
        for (int c = 0; c < HDIM; ++c) o[c] = 0.f;
#pragma unroll
        for (int kk = 0; kk < 25; ++kk) {
            const float wk = lg[kk] * inv;
            const int di = kk / 5 - 2, dj = kk % 5 - 2;
            const _Float16* vp = vt + ((ly + di) * 20 + (lx + dj)) * KSTR;
#pragma unroll
            for (int c8 = 0; c8 < 8; ++c8) {
                v8h vv = *(const v8h*)(vp + c8 * 8);
#pragma unroll
                for (int j = 0; j < 8; ++j) o[c8 * 8 + j] += wk * (float)vv[j];
            }
        }
        _Float16* op = ah + (size_t)mypix * 256 + hd * HDIM;
#pragma unroll
        for (int c = 0; c < HDIM; ++c) op[c] = (_Float16)o[c];
    }
}

// ---------------------------------------------------------------------------
// K3: output projection (ah @ Wp) with WMMA; Wp staged via TDM into LDS, and
// a per-wave LDS transpose so f32 stores to channel-major [T,C,H,W] are
// coalesced along W.
// ---------------------------------------------------------------------------
__global__ void __launch_bounds__(256) k_proj(const _Float16* __restrict__ ah,
                                              const _Float16* __restrict__ wt,
                                              float* __restrict__ out) {
    __shared__ _Float16 wlds[256 * WPITCH];   // 135,168 B
    __shared__ float tb[8 * 16 * 17];         // per-wave 16x16 tile, pitch 17
    const int tid  = threadIdx.x;
    const int lane = tid & 31;
    const int wave = tid >> 5;
    float* myt = tb + wave * 272;

    const int rt = blockIdx.x * 8 + wave;
    const int pixbase = rt * 16;
    const int t  = pixbase >> 16;
    const int h  = (pixbase >> 8) & 255;
    const int wb = pixbase & 255;
    const int ln  = lane & 15;
    const int ak8 = (lane >> 4) << 3;
    const int bk16 = (lane >> 4) << 3;

    stage_weight_lds(wlds, wt + 3 * 65536, wave, tid);   // Wp

    v16h afr[8];
    {
        const _Float16* arow = ah + (size_t)(pixbase + ln) * 256 + ak8;
#pragma unroll
        for (int ks = 0; ks < 8; ++ks) {
            v8h lo = *(const v8h*)(arow + ks * 32);
            v8h hi = *(const v8h*)(arow + ks * 32 + 16);
            afr[ks] = cat16(lo, hi);
        }
    }

    for (int ct = 0; ct < 16; ++ct) {
        v8f acc = {};
        const _Float16* wrow = wlds + (ct * 16 + ln) * WPITCH + bk16;
#pragma unroll
        for (int ks = 0; ks < 8; ++ks) {
            v8h lo = *(const v8h*)(wrow + ks * 32);
            v8h hi = *(const v8h*)(wrow + ks * 32 + 16);
            v16h b = cat16(lo, hi);
            acc = __builtin_amdgcn_wmma_f32_16x16x32_f16(
                    false, afr[ks], false, b, (short)0, acc, false, false);
        }
        // Y[m][n] -> LDS (same-wave region; DS ops are in-order per wave).
#pragma unroll
        for (int r = 0; r < 8; ++r)
            myt[(r + ak8) * 17 + ln] = acc[r];
        // Read transposed: lanes sweep W for a fixed channel -> coalesced.
        const int c0 = ct * 16;
#pragma unroll
        for (int cc = 0; cc < 8; ++cc) {
            const int c = c0 + cc * 2 + (lane >> 4);
            const int wcol = wb + ln;
            out[(size_t)(t * CC + c) * HW + h * WWD + wcol] =
                myt[ln * 17 + cc * 2 + (lane >> 4)];
        }
    }
}

// ---------------------------------------------------------------------------
extern "C" void kernel_launch(void* const* d_in, const int* in_sizes, int n_in,
                              void* d_out, int out_size, void* d_ws, size_t ws_size,
                              hipStream_t stream) {
    const float* x    = (const float*)d_in[0];
    const float* sims = (const float*)d_in[1];
    // d_in[2] = flows (all zeros, unused by the reference math)
    const float* Wq = (const float*)d_in[3];
    const float* Wk = (const float*)d_in[4];
    const float* Wv = (const float*)d_in[5];
    const float* Wp = (const float*)d_in[6];
    float* out = (float*)d_out;

    // Workspace layout (f16): Wt[4*256*256] | qh | kh | vh | ah  (~512.5 MB)
    _Float16* wt = (_Float16*)d_ws;
    _Float16* qh = wt + 4 * 65536;
    _Float16* kh = qh + (size_t)NPIX * 256;
    _Float16* vh = kh + (size_t)NPIX * 256;
    _Float16* ah = vh + (size_t)NPIX * 256;

    k_wt  <<<dim3(256, 4),      256, 0, stream>>>(Wq, Wk, Wv, Wp, wt);
    k_qkv <<<dim3(NPIX / 128),  256, 0, stream>>>(x, wt, qh, kh, vh);
    k_attn<<<dim3(16, 16, TT),  256, 0, stream>>>(qh, kh, vh, sims, ah);
    k_proj<<<dim3(NPIX / 128),  256, 0, stream>>>(ah, wt, out);
}
